// Quantization_71004399337674
// MI455X (gfx1250) — compile-verified, matrix operand order checked by reading, not run on hardware
//
#include <hip/hip_runtime.h>

#define NTOK   16384          // B*S
#define DMODEL 512
#define GV     640            // G*V
#define G_     2
#define V_     320
#define CVD    128
#define NTILES 40             // GV/16
#define KSTEPS 16             // DMODEL/32

typedef __attribute__((ext_vector_type(16))) __bf16 v16bf;
typedef __attribute__((ext_vector_type(8)))  float  v8f;

union Frag16 { v16bf v; uint4 q[2]; unsigned short s[16]; };

__device__ __forceinline__ unsigned short f2bf(float f) {
    unsigned int u = __float_as_uint(f);
    u += 0x7fffu + ((u >> 16) & 1u);      // round-to-nearest-even
    return (unsigned short)(u >> 16);
}

// ---------------------------------------------------------------------------
// k0: zero marginal accumulator; repack W (f32, [512][640]) into bf16 laid out
// in exact WMMA B-fragment order: Bpack[t][kt][lane][e] (e = 16 bf16 = 32B).
// ---------------------------------------------------------------------------
__global__ void k0_pack(const float* __restrict__ W,
                        float* __restrict__ sums,
                        unsigned short* __restrict__ Bpack) {
    int gid = blockIdx.x * blockDim.x + threadIdx.x;
    if (gid < GV) sums[gid] = 0.0f;
    int j = gid - GV;
    if (j >= 0 && j < NTILES * KSTEPS * 32 * 16) {
        int e    = j & 15;
        int lane = (j >> 4) & 31;
        int kt   = (j >> 9) & 15;
        int t    = j >> 13;
        int n    = t * 16 + (lane & 15);
        int off  = (lane >= 16) ? 8 : 0;
        int k    = kt * 32 + off + ((e < 8) ? e : e + 8);
        Bpack[j] = f2bf(W[k * GV + n]);
    }
}

// ---------------------------------------------------------------------------
// k1: fused GEMM (bf16 WMMA) + gumbel argmax + codebook gather + marginal acc.
// One block = 16 tokens, 8 waves; each wave owns 5 of the 40 N-tiles.
// ---------------------------------------------------------------------------
__global__ __launch_bounds__(256)
void k1_main(const float* __restrict__ hid, const float* __restrict__ bias,
             const float* __restrict__ cv,  const float* __restrict__ gmb,
             const unsigned short* __restrict__ Bpack,
             float* __restrict__ out, float* __restrict__ sums) {
    __shared__ unsigned short As[16][DMODEL + 8];  // bf16 A tile, padded rows
    __shared__ float Lg[16][GV];                   // logits tile
    __shared__ float Mp[GV];                       // per-block marginal partial

    const int tid  = threadIdx.x;
    const int lane = tid & 31;
    const int w    = tid >> 5;
    const int rb   = blockIdx.x;

    // prefetch this block's gumbel slab (16 tokens * 640 f32 = 40 KB) so the
    // HBM fetch overlaps the whole WMMA phase (global_prefetch_b8)
    {
        const char* gbase = (const char*)(gmb + (size_t)(rb * 16) * GV);
        for (int i = tid; i < (16 * GV * 4) / 128; i += 256)
            __builtin_prefetch(gbase + i * 128, 0, 3);
    }

    for (int i = tid; i < GV; i += 256) Mp[i] = 0.0f;

    // stage A (16 x 512 f32 -> bf16 LDS), coalesced
    for (int f = tid; f < 16 * DMODEL; f += 256) {
        int r = f >> 9, c = f & 511;
        As[r][c] = f2bf(hid[(size_t)(rb * 16 + r) * DMODEL + c]);
    }
    __syncthreads();

    // ---- GEMM phase ----
    const int mrow = lane & 15;
    const int off  = (lane >= 16) ? 8 : 0;

    // per-lane invariant base pointers -> all loads use immediate offsets
    const char* aB = (const char*)&As[0][0] + mrow * ((DMODEL + 8) * 2) + off * 2;
    const char* bB = (const char*)Bpack + (size_t)w * (5 * KSTEPS * 32 * 32)
                                        + (size_t)lane * 32;

    v8f acc[5];
#pragma unroll
    for (int i = 0; i < 5; ++i) {
        int t = w * 5 + i;
        float bb = bias[t * 16 + (lane & 15)];     // bias depends on N only
        acc[i] = (v8f){bb, bb, bb, bb, bb, bb, bb, bb};
    }

#pragma unroll 4
    for (int kt = 0; kt < KSTEPS; ++kt) {
        Frag16 a;
        a.q[0] = *(const uint4*)(aB + kt * 64);        // K = k0..k0+7
        a.q[1] = *(const uint4*)(aB + kt * 64 + 32);   // K = k0+16..k0+23
#pragma unroll
        for (int i = 0; i < 5; ++i) {
            Frag16 b;
            b.q[0] = *(const uint4*)(bB + i * 16384 + kt * 1024);
            b.q[1] = *(const uint4*)(bB + i * 16384 + kt * 1024 + 16);
            acc[i] = __builtin_amdgcn_wmma_f32_16x16x32_bf16(
                false, a.v, false, b.v, (short)0, acc[i], false, false);
        }
    }

    // scatter C tiles to LDS logits: VGPR r <-> M = r + 8*(lane>=16), N = lane&15
#pragma unroll
    for (int i = 0; i < 5; ++i) {
        int t = w * 5 + i;
        int n = t * 16 + (lane & 15);
        int mb = (lane >> 4) << 3;
#pragma unroll
        for (int r = 0; r < 8; ++r) Lg[mb + r][n] = acc[i][r];
    }
    __syncthreads();

    // ---- softmax / argmax / gather phase: 32 (token,group) tasks, 8 waves ----
    for (int j = 0; j < 4; ++j) {
        int task = w + 8 * j;
        int m = task >> 1, g = task & 1;
        int n = rb * 16 + m;
        const float* Lrow = &Lg[m][g * V_];
        const float* gp   = gmb + (size_t)(n * G_ + g) * V_;

        float lv[10];
        float maxS = -3.4e38f, maxN = -3.4e38f;
        int idx = 0;
#pragma unroll
        for (int q = 0; q < 10; ++q) {
            int v = lane + 32 * q;
            float l = Lrow[v];
            lv[q] = l;
            float nz = l + gp[v];
            if (nz > maxN) { maxN = nz; idx = v; }
            maxS = fmaxf(maxS, l);
        }
#pragma unroll
        for (int o = 16; o > 0; o >>= 1) {
            float om = __shfl_xor(maxN, o, 32);
            int   oi = __shfl_xor(idx,  o, 32);
            if (om > maxN || (om == maxN && oi < idx)) { maxN = om; idx = oi; }
            maxS = fmaxf(maxS, __shfl_xor(maxS, o, 32));
        }

        float ev[10], se = 0.0f;
#pragma unroll
        for (int q = 0; q < 10; ++q) { ev[q] = __expf(lv[q] - maxS); se += ev[q]; }
#pragma unroll
        for (int o = 16; o > 0; o >>= 1) se += __shfl_xor(se, o, 32);
        float inv = 1.0f / se;
#pragma unroll
        for (int q = 0; q < 10; ++q)
            atomicAdd(&Mp[g * V_ + lane + 32 * q], ev[q] * inv);

        // hard one-hot == direct codevector gather; 128 floats, float4/lane
        const float4* cvr = (const float4*)(cv + ((size_t)g * V_ + idx) * CVD);
        float4 c = cvr[lane];
        *(float4*)(out + (size_t)n * (G_ * CVD) + g * CVD + lane * 4) = c;
    }
    __syncthreads();

    // flush block-partial marginal (640 global f32 atomics per block)
    for (int i = tid; i < GV; i += 256) atomicAdd(&sums[i], Mp[i]);
}

// ---------------------------------------------------------------------------
// k2: perplexity = sum_g exp(-sum_v m*log(m+1e-7)), m = sums/N
// ---------------------------------------------------------------------------
__global__ void k2_fin(const float* __restrict__ sums, float* __restrict__ outp) {
    __shared__ float sh[GV];
    int t = threadIdx.x;
    for (int i = t; i < GV; i += blockDim.x) {
        float m = sums[i] * (1.0f / (float)NTOK);
        sh[i] = m * __logf(m + 1e-7f);
    }
    __syncthreads();
    if (t == 0) {
        float p = 0.0f;
        for (int g = 0; g < G_; ++g) {
            float e = 0.0f;
            for (int v = 0; v < V_; ++v) e += sh[g * V_ + v];
            p += __expf(-e);
        }
        outp[0] = p;
    }
}

// ---------------------------------------------------------------------------
extern "C" void kernel_launch(void* const* d_in, const int* in_sizes, int n_in,
                              void* d_out, int out_size, void* d_ws, size_t ws_size,
                              hipStream_t stream) {
    const float* hid = (const float*)d_in[0];   // (8,2048,512)
    const float* W   = (const float*)d_in[1];   // (512,640)
    const float* b   = (const float*)d_in[2];   // (640,)
    const float* cv  = (const float*)d_in[3];   // (1,640,128)
    const float* gmb = (const float*)d_in[4];   // (32768,320)
    float* out  = (float*)d_out;                // 16384*256 floats + 1 scalar
    float* sums = (float*)d_ws;                 // 640 f32 marginal accumulator
    unsigned short* Bpack = (unsigned short*)((char*)d_ws + 4096); // 640 KB

    int total0 = GV + NTILES * KSTEPS * 32 * 16;
    k0_pack<<<(total0 + 255) / 256, 256, 0, stream>>>(W, sums, Bpack);
    k1_main<<<NTOK / 16, 256, 0, stream>>>(hid, b, cv, gmb, Bpack, out, sums);
    k2_fin<<<1, 256, 0, stream>>>(sums, out + (size_t)NTOK * G_ * CVD);
}
